// QuantizedLinear_4140348473394
// MI455X (gfx1250) — matrix operand order repaired and checked by reading, not compiled, not run on hardware
//
#include <hip/hip_runtime.h>

typedef __attribute__((ext_vector_type(16))) _Float16 v16h;
typedef __attribute__((ext_vector_type(8)))  _Float16 v8h;
typedef __attribute__((ext_vector_type(4)))  _Float16 v4h;
typedef __attribute__((ext_vector_type(8)))  float    v8f;

#define M_TOTAL 4096
#define N_TOTAL 11008
#define K_TOTAL 4096
#define KB4     (K_TOTAL / 4)   // int32 elements per output row (1 byte per int32)
#define BM 128
#define BN 128
#define BK 64
#define LDSS 72                 // padded LDS row stride in halves (144 B = 36 banks)

__global__ __launch_bounds__(256)
void ternary_wmma_gemm(const float* __restrict__ x,
                       const int*   __restrict__ pw,
                       const float* __restrict__ scale_p,
                       float*       __restrict__ out) {
  __shared__ _Float16 Ash[BM * LDSS];
  __shared__ _Float16 Bsh[BN * LDSS];

  const int tid   = threadIdx.x;
  const int lane  = tid & 31;
  const int wave  = tid >> 5;
  const int l16   = lane & 15;
  const bool hi16 = lane >= 16;

  const int mBase = blockIdx.y * BM;
  const int nBase = blockIdx.x * BN;

  // 8 waves: 4 along M (32 rows each) x 2 along N (64 cols each)
  const int wm = (wave & 3) * 32;
  const int wn = (wave >> 2) * 64;

  v8f acc[2][4];
#pragma unroll
  for (int i = 0; i < 2; ++i)
#pragma unroll
    for (int j = 0; j < 4; ++j)
      acc[i][j] = (v8f)(0.0f);

  // ---- staging assignments ----
  // A: 128 rows x 16 float4/row = 2048 float4 -> 8 per thread
  // B: 128 rows x 4 int4/row    =  512 int4   -> 2 per thread
  // (int4 = 4 bytes of packed weight = 16 K-values)
  float4 aReg[8];
  int4   bReg[2];

  auto load_tile = [&](int k0) {
#pragma unroll
    for (int r = 0; r < 8; ++r) {
      const int f   = tid + r * 256;
      const int row = f >> 4;
      const int c4  = f & 15;
      aReg[r] = *(const float4*)(x + (size_t)(mBase + row) * K_TOTAL + k0 + c4 * 4);
    }
#pragma unroll
    for (int r = 0; r < 2; ++r) {
      const int id = tid + r * 256;
      const int n  = id >> 2;
      const int q  = id & 3;          // k-chunk of 16 values
      bReg[r] = *(const int4*)(pw + (size_t)(nBase + n) * KB4 + (k0 >> 2) + q * 4);
    }
  };

  auto store_tile = [&]() {
#pragma unroll
    for (int r = 0; r < 8; ++r) {
      const int f   = tid + r * 256;
      const int row = f >> 4;
      const int c4  = f & 15;
      v4h h;
      h[0] = (_Float16)aReg[r].x;
      h[1] = (_Float16)aReg[r].y;
      h[2] = (_Float16)aReg[r].z;
      h[3] = (_Float16)aReg[r].w;
      *(v4h*)(&Ash[row * LDSS + c4 * 4]) = h;
    }
#pragma unroll
    for (int r = 0; r < 2; ++r) {
      const int id = tid + r * 256;
      const int n  = id >> 2;
      const int q  = id & 3;
      const int b0 = bReg[r].x, b1 = bReg[r].y, b2 = bReg[r].z, b3 = bReg[r].w;
      v8h lo, hi;
#pragma unroll
      for (int j = 0; j < 4; ++j) {
        lo[j]     = (_Float16)(float)(((b0 >> (2 * j)) & 3) - 1);
        lo[4 + j] = (_Float16)(float)(((b1 >> (2 * j)) & 3) - 1);
        hi[j]     = (_Float16)(float)(((b2 >> (2 * j)) & 3) - 1);
        hi[4 + j] = (_Float16)(float)(((b3 >> (2 * j)) & 3) - 1);
      }
      _Float16* dst = &Bsh[n * LDSS + q * 16];
      *(v8h*)(dst)     = lo;
      *(v8h*)(dst + 8) = hi;
    }
  };

  load_tile(0);  // prologue prefetch

  for (int k0 = 0; k0 < K_TOTAL; k0 += BK) {
    store_tile();
    __syncthreads();

    // software pipeline: issue next tile's global loads before compute
    if (k0 + BK < K_TOTAL) load_tile(k0 + BK);

    // ---- 2 K-substeps of 32, 8 WMMAs each ----
#pragma unroll
    for (int kk = 0; kk < BK; kk += 32) {
      v16h af[2];
#pragma unroll
      for (int i = 0; i < 2; ++i) {
        const _Float16* ap =
            &Ash[(wm + i * 16 + l16) * LDSS + kk + (hi16 ? 16 : 0)];
        v8h lo = *(const v8h*)(ap);       // K = base .. base+7
        v8h hi = *(const v8h*)(ap + 8);   // K = base+8 .. base+15
        af[i] = __builtin_shufflevector(lo, hi, 0, 1, 2, 3, 4, 5, 6, 7, 8, 9,
                                        10, 11, 12, 13, 14, 15);
      }
      v16h bf[4];
#pragma unroll
      for (int j = 0; j < 4; ++j) {
        const _Float16* bp =
            &Bsh[(wn + j * 16 + l16) * LDSS + kk + (hi16 ? 16 : 0)];
        v8h lo = *(const v8h*)(bp);       // K = base .. base+7
        v8h hi = *(const v8h*)(bp + 8);   // K = base+8 .. base+15
        bf[j] = __builtin_shufflevector(lo, hi, 0, 1, 2, 3, 4, 5, 6, 7, 8, 9,
                                        10, 11, 12, 13, 14, 15);
      }
#pragma unroll
      for (int i = 0; i < 2; ++i)
#pragma unroll
        for (int j = 0; j < 4; ++j)
          acc[i][j] = __builtin_amdgcn_wmma_f32_16x16x32_f16(
              false, af[i], false, bf[j], (short)0, acc[i][j], false, false);
    }

    __syncthreads();
  }

  // ---- epilogue: scale once, non-temporal fp32 stores ----
  // (write-once 180 MB stream must not evict the L2-resident x / packed-W
  //  working set; TH=NT keeps L2 for the reused operands)
  const float s    = scale_p[0];
  const int   mAdd = hi16 ? 8 : 0;
#pragma unroll
  for (int i = 0; i < 2; ++i) {
    const int mRow = mBase + wm + i * 16 + mAdd;
#pragma unroll
    for (int j = 0; j < 4; ++j) {
      const int col = nBase + wn + j * 16 + l16;
      float* op = out + (size_t)mRow * N_TOTAL + col;
#pragma unroll
      for (int v = 0; v < 8; ++v)
        __builtin_nontemporal_store(s * acc[i][j][v], op + (size_t)v * N_TOTAL);
    }
  }
}

extern "C" void kernel_launch(void* const* d_in, const int* in_sizes, int n_in,
                              void* d_out, int out_size, void* d_ws, size_t ws_size,
                              hipStream_t stream) {
  const float* x   = (const float*)d_in[0];
  const int*   pw  = (const int*)d_in[1];
  const float* sc  = (const float*)d_in[2];
  float*       out = (float*)d_out;

  dim3 grid(N_TOTAL / BN, M_TOTAL / BM);  // (86, 32)
  dim3 block(256);
  ternary_wmma_gemm<<<grid, block, 0, stream>>>(x, pw, sc, out);
}
